// AttentionBlock_86818468922172
// MI455X (gfx1250) — matrix operand order repaired and checked by reading, not compiled, not run on hardware
//
#include <hip/hip_runtime.h>

#define CC 256
#define NN 4096
#define NH 4
#define DH 64
#define NGROUPS 32

typedef _Float16 v16h __attribute__((ext_vector_type(16)));
typedef _Float16 v8h  __attribute__((ext_vector_type(8)));
typedef float    v8f  __attribute__((ext_vector_type(8)));

#define DEV __device__ __forceinline__

DEV v16h cat8(v8h lo, v8h up) {
  v16h r;
  #pragma unroll
  for (int i = 0; i < 8; ++i) { r[i] = lo[i]; r[i + 8] = up[i]; }
  return r;
}

// A fragment (16x32 f16): elems 0..15 = K kbase+16*hi+i (contiguous)
DEV v16h load_a16(const _Float16* rowptr, int kbase, int hi) {
  v8h lo = *reinterpret_cast<const v8h*>(rowptr + kbase + 16 * hi);
  v8h up = *reinterpret_cast<const v8h*>(rowptr + kbase + 16 * hi + 8);
  return cat8(lo, up);
}

// B fragment (32x16 f16): lane holds 16 contiguous K values (K0 = 16*hi) for col lane%16
DEV v16h load_b16(const _Float16* p) {
  v8h lo = *reinterpret_cast<const v8h*>(p);
  v8h up = *reinterpret_cast<const v8h*>(p + 8);
  return cat8(lo, up);
}

DEV v8f zero8() {
  v8f z;
  #pragma unroll
  for (int i = 0; i < 8; ++i) z[i] = 0.0f;
  return z;
}

DEV v8f wmma_f16(v16h a, v16h b, v8f c) {
  return __builtin_amdgcn_wmma_f32_16x16x32_f16(false, a, false, b, (short)0, c, false, false);
}

// reduce across each 16-lane group (rows of the C/D tile live in half-wave groups)
DEV float rowmax16(float v) {
  #pragma unroll
  for (int m = 1; m <= 8; m <<= 1) v = fmaxf(v, __shfl_xor(v, m, 32));
  return v;
}
DEV float rowsum16(float v) {
  #pragma unroll
  for (int m = 1; m <= 8; m <<= 1) v += __shfl_xor(v, m, 32);
  return v;
}

// ---------------- GroupNorm: x[B,C,N] f32 -> h_t[B,N,C] f16 ----------------
__global__ void gn_kernel(const float* __restrict__ x, const float* __restrict__ gw,
                          const float* __restrict__ gb, _Float16* __restrict__ h_t) {
  const int b = blockIdx.x / NGROUPS;
  const int g = blockIdx.x % NGROUPS;
  const int CPG = CC / NGROUPS;                 // 8 channels per group
  const size_t base = ((size_t)b * CC + (size_t)g * CPG) * NN;
  const int total = CPG * NN;                   // 32768
  float s = 0.f, ss = 0.f;
  for (int e = threadIdx.x; e < total; e += blockDim.x) {
    float v = x[base + e];
    s += v; ss += v * v;
  }
  __shared__ float rs[256], rq[256];
  rs[threadIdx.x] = s; rq[threadIdx.x] = ss;
  __syncthreads();
  for (int off = 128; off > 0; off >>= 1) {
    if ((int)threadIdx.x < off) { rs[threadIdx.x] += rs[threadIdx.x + off]; rq[threadIdx.x] += rq[threadIdx.x + off]; }
    __syncthreads();
  }
  const float inv_n = 1.0f / (float)total;
  const float mean = rs[0] * inv_n;
  const float var  = rq[0] * inv_n - mean * mean;
  const float rstd = rsqrtf(var + 1e-5f);
  for (int e = threadIdx.x; e < total; e += blockDim.x) {
    int cl = e >> 12;                // e / N
    int n  = e & (NN - 1);
    int c  = g * CPG + cl;
    float v = (x[base + e] - mean) * rstd * gw[c] + gb[c];
    h_t[((size_t)b * NN + n) * CC + c] = (_Float16)v;
  }
}

__global__ void f32_to_f16_kernel(const float* __restrict__ in, _Float16* __restrict__ out, int n) {
  int i = blockIdx.x * 256 + threadIdx.x;
  if (i < n) out[i] = (_Float16)in[i];
}

// --------- QKV GEMM: [3C,C] x h_t -> q_t/k_t [B,h,N,64], v_s [B,h,64,N] ----------
__global__ void qkv_gemm_kernel(const _Float16* __restrict__ W, const float* __restrict__ bias,
                                const _Float16* __restrict__ h_t,
                                _Float16* __restrict__ q_t, _Float16* __restrict__ k_t,
                                _Float16* __restrict__ v_s) {
  const int lane = threadIdx.x & 31;
  const int wave = threadIdx.x >> 5;
  const int hi = lane >> 4;
  const int lo = lane & 15;
  const int b = blockIdx.z;
  const int otile = blockIdx.y * 16;
  const int n = (blockIdx.x * 4 + wave) * 16 + lo;

  const _Float16* wrow = W + (size_t)(otile + lo) * CC;
  const _Float16* hrow = h_t + ((size_t)b * NN + n) * CC + 16 * hi;

  v8f acc = zero8();
  #pragma unroll
  for (int kb = 0; kb < CC; kb += 32) {
    v16h a  = load_a16(wrow, kb, hi);
    v16h bb = load_b16(hrow + kb);
    acc = wmma_f16(a, bb, acc);
  }

  const int obase  = otile + 8 * hi;        // first output row of this lane's 8 elems
  const int region = otile >> 8;            // 0=q 1=k 2=v  (uniform per wave)
  const int oo     = obase & 255;
  const int head   = oo >> 6;
  const int c0     = oo & 63;
  const size_t bh  = (size_t)(b * NH + head);

  if (region < 2) {
    v8h pack;
    #pragma unroll
    for (int j = 0; j < 8; ++j) pack[j] = (_Float16)(acc[j] + bias[obase + j]);
    _Float16* dst = (region == 0 ? q_t : k_t) + (bh * NN + n) * DH + c0;
    *reinterpret_cast<v8h*>(dst) = pack;
  } else {
    #pragma unroll
    for (int j = 0; j < 8; ++j)
      v_s[(bh * DH + c0 + j) * NN + n] = (_Float16)(acc[j] + bias[obase + j]);
  }
}

// ---------------- Flash attention: one 16-query tile per wave ----------------
__global__ void flash_attn_kernel(const _Float16* __restrict__ q_t, const _Float16* __restrict__ k_t,
                                  const _Float16* __restrict__ v_s, _Float16* __restrict__ out_t) {
  const int lane = threadIdx.x & 31;
  const int wave = threadIdx.x >> 5;
  const int hi = lane >> 4;
  const int lo = lane & 15;
  const int b = blockIdx.z;
  const int head = blockIdx.y;
  const int qtile = (blockIdx.x * 4 + wave) * 16;
  const size_t bh = (size_t)(b * NH + head);
  const _Float16* qm = q_t + bh * (size_t)NN * DH;
  const _Float16* km = k_t + bh * (size_t)NN * DH;
  const _Float16* vm = v_s + bh * (size_t)DH * NN;

  __shared__ float P[4][16][32];             // per-wave P tile (queries x 32 keys)

  const _Float16* qrow = qm + (size_t)(qtile + lo) * DH;
  const v16h a0 = load_a16(qrow, 0, hi);     // K = d 0..31
  const v16h a1 = load_a16(qrow, 32, hi);    // K = d 32..63

  float mrun[8], lrun[8];
  v8f acc[4];
  #pragma unroll
  for (int j = 0; j < 8; ++j) { mrun[j] = -3.0e38f; lrun[j] = 0.f; }
  #pragma unroll
  for (int t = 0; t < 4; ++t) acc[t] = zero8();

  const float scale = 0.125f;                // 64^-0.5

  for (int kb = 0; kb < NN; kb += 32) {
    {
      int nk = (kb + 32) & (NN - 1);
      __builtin_prefetch((const void*)(km + (size_t)(nk + lo) * DH), 0, 1);
    }
    const _Float16* k0p = km + (size_t)(kb + lo) * DH + 16 * hi;
    const _Float16* k1p = km + (size_t)(kb + 16 + lo) * DH + 16 * hi;
    v16h b00 = load_b16(k0p);
    v16h b01 = load_b16(k0p + 32);
    v16h b10 = load_b16(k1p);
    v16h b11 = load_b16(k1p + 32);
    v8f s0 = zero8();
    s0 = wmma_f16(a0, b00, s0);
    s0 = wmma_f16(a1, b01, s0);
    v8f s1 = zero8();
    s1 = wmma_f16(a0, b10, s1);
    s1 = wmma_f16(a1, b11, s1);

    float corr[8];
    #pragma unroll
    for (int j = 0; j < 8; ++j) {
      float x0 = s0[j] * scale;
      float x1 = s1[j] * scale;
      float mx = rowmax16(fmaxf(x0, x1));
      float mnew = fmaxf(mrun[j], mx);
      float p0 = __expf(x0 - mnew);
      float p1 = __expf(x1 - mnew);
      corr[j] = __expf(mrun[j] - mnew);
      lrun[j] = lrun[j] * corr[j] + rowsum16(p0 + p1);
      mrun[j] = mnew;
      P[wave][j + 8 * hi][lo]      = p0;     // D-layout -> row-major P in LDS
      P[wave][j + 8 * hi][lo + 16] = p1;
    }
    #pragma unroll
    for (int t = 0; t < 4; ++t)
      #pragma unroll
      for (int j = 0; j < 8; ++j) acc[t][j] *= corr[j];

    __syncthreads();
    v16h pf;                                  // re-read P as A fragment (16x32, K=keys)
    #pragma unroll
    for (int i = 0; i < 8; ++i) {
      pf[i]     = (_Float16)P[wave][lo][16 * hi + i];
      pf[i + 8] = (_Float16)P[wave][lo][16 * hi + 8 + i];
    }
    __syncthreads();

    #pragma unroll
    for (int t = 0; t < 4; ++t) {
      v16h bv = load_b16(vm + (size_t)(t * 16 + lo) * NN + kb + 16 * hi);
      acc[t] = wmma_f16(pf, bv, acc[t]);
    }
  }

  float inv[8];
  #pragma unroll
  for (int j = 0; j < 8; ++j) inv[j] = 1.0f / lrun[j];
  #pragma unroll
  for (int t = 0; t < 4; ++t)
    #pragma unroll
    for (int j = 0; j < 8; ++j)
      out_t[((size_t)b * NN + qtile + j + 8 * hi) * CC + head * DH + t * 16 + lo] =
          (_Float16)(acc[t][j] * inv[j]);
}

// ---------------- Proj GEMM + bias + residual -> d_out f32 [B,C,N] ----------------
__global__ void proj_gemm_kernel(const _Float16* __restrict__ W, const float* __restrict__ bias,
                                 const _Float16* __restrict__ out_t, const float* __restrict__ x,
                                 float* __restrict__ y) {
  const int lane = threadIdx.x & 31;
  const int wave = threadIdx.x >> 5;
  const int hi = lane >> 4;
  const int lo = lane & 15;
  const int b = blockIdx.z;
  const int otile = blockIdx.y * 16;
  const int n = (blockIdx.x * 4 + wave) * 16 + lo;

  const _Float16* wrow = W + (size_t)(otile + lo) * CC;
  const _Float16* orow = out_t + ((size_t)b * NN + n) * CC + 16 * hi;

  v8f acc = zero8();
  #pragma unroll
  for (int kb = 0; kb < CC; kb += 32) {
    v16h a  = load_a16(wrow, kb, hi);
    v16h bb = load_b16(orow + kb);
    acc = wmma_f16(a, bb, acc);
  }
  #pragma unroll
  for (int j = 0; j < 8; ++j) {
    int o = otile + 8 * hi + j;
    size_t idx = ((size_t)b * CC + o) * NN + n;
    y[idx] = acc[j] + bias[o] + x[idx];
  }
}

extern "C" void kernel_launch(void* const* d_in, const int* in_sizes, int n_in,
                              void* d_out, int out_size, void* d_ws, size_t ws_size,
                              hipStream_t stream) {
  (void)n_in; (void)out_size; (void)ws_size;
  const float* x      = (const float*)d_in[0];
  // d_in[1] = t_emb (unused by the reference)
  const float* gn_w   = (const float*)d_in[2];
  const float* gn_b   = (const float*)d_in[3];
  const float* qkv_w  = (const float*)d_in[4];
  const float* qkv_b  = (const float*)d_in[5];
  const float* proj_w = (const float*)d_in[6];
  const float* proj_b = (const float*)d_in[7];
  float* y = (float*)d_out;
  const int B = in_sizes[0] / (CC * NN);

  char* p = (char*)d_ws;
  auto take = [&](size_t bytes) { char* r = p; p += (bytes + 255) & ~(size_t)255; return r; };
  _Float16* h_t   = (_Float16*)take((size_t)B * NN * CC * 2);
  _Float16* q_t   = (_Float16*)take((size_t)B * NH * NN * DH * 2);
  _Float16* k_t   = (_Float16*)take((size_t)B * NH * NN * DH * 2);
  _Float16* v_s   = (_Float16*)take((size_t)B * NH * DH * NN * 2);
  _Float16* out_t = (_Float16*)take((size_t)B * NN * CC * 2);
  _Float16* wq16  = (_Float16*)take((size_t)3 * CC * CC * 2);
  _Float16* wp16  = (_Float16*)take((size_t)CC * CC * 2);

  f32_to_f16_kernel<<<dim3((3 * CC * CC + 255) / 256), 256, 0, stream>>>(qkv_w, wq16, 3 * CC * CC);
  f32_to_f16_kernel<<<dim3((CC * CC + 255) / 256), 256, 0, stream>>>(proj_w, wp16, CC * CC);
  gn_kernel<<<dim3(B * NGROUPS), 256, 0, stream>>>(x, gn_w, gn_b, h_t);
  qkv_gemm_kernel<<<dim3(NN / 64, (3 * CC) / 16, B), 128, 0, stream>>>(wq16, qkv_b, h_t, q_t, k_t, v_s);
  flash_attn_kernel<<<dim3((NN / 16) / 4, NH, B), 128, 0, stream>>>(q_t, k_t, v_s, out_t);
  proj_gemm_kernel<<<dim3(NN / 64, CC / 16, B), 128, 0, stream>>>(wp16, proj_b, out_t, x, y);
}
